// MoE_56032143344235
// MI455X (gfx1250) — matrix operand order repaired and checked by reading, not compiled, not run on hardware
//
#include <hip/hip_runtime.h>

// Fused dense-MoE forward for MI455X (gfx1250, wave32, WMMA).
//   B=8, T=4096, D=1024, E=8, H=32 (2H=64), N = B*T = 32768 tokens.
// One wave owns a 16-token tile end-to-end:
//   router (WMMA) -> softmax -> per-expert up-proj (WMMA) -> SwiGLU*prob
//   -> down-proj + prob-weighted bias (WMMA) -> out.
// Every operand lives in LDS in WMMA fragment register-image order (each
// lane's 8 fragment dwords contiguous), so ALL fragment loads are 16/32-byte
// LDS accesses; zero-padded fragments use a pointer-select into a zero buffer
// instead of per-element masking.  After phase 2 the x/W1 LDS (256KB) is
// repurposed as an f16 W2 fragment cache (2 halves of 512 columns).

#define Dm     1024
#define Em     8
#define Hm     32
#define H2m    64
#define TILE_M 16
#define NWAVES 4
#define NTOK   (8 * 4096)
#define KC     (Dm / 32)   // 32 K-chunks of 32

typedef __attribute__((ext_vector_type(16))) _Float16 v16h;
typedef __attribute__((ext_vector_type(8)))  _Float16 v8h;
typedef __attribute__((ext_vector_type(4)))  _Float16 v4h;
typedef __attribute__((ext_vector_type(8)))  float    v8f;
typedef __attribute__((ext_vector_type(4)))  float    v4f;

// LDS fragment-image layout = [fragment][lane 0..31][16 halves]
#define SA_HALVES   (KC * 32 * 16)            // x tile per wave: 32 frags
#define SW_HALVES   (KC * 4 * 32 * 16)        // W1[e]: 32 kc x 4 nt frags
#define SACT_HALVES (Em * 32 * 16)            // act per wave: 8 frags

#define SA_BYTES   (NWAVES * SA_HALVES * 2)   // 131072
#define SW_BYTES   (SW_HALVES * 2)            // 131072  (sA+sW reused as W2 cache)
#define SACT_BYTES (NWAVES * SACT_HALVES * 2) // 32768
#define SP_BYTES   (NWAVES * TILE_M * Em * 4) // 2048
#define SB2_BYTES  (Dm * Em * 2)              // 16384  b2 strip [col][k] f16
#define SWR_BYTES  (Dm * Em * 2)              // 16384  Wr strip [k>>4][n][k&15] f16
#define SZ_BYTES   32                         // zero buffer (one 32B fragment)

#define SACT_OFF (SA_BYTES + SW_BYTES)
#define SP_OFF   (SACT_OFF + SACT_BYTES)
#define SB2_OFF  (SP_OFF + SP_BYTES)
#define SWR_OFF  (SB2_OFF + SB2_BYTES)
#define SZ_OFF   (SWR_OFF + SWR_BYTES)
#define LDS_TOTAL (SZ_OFF + SZ_BYTES)         // 329760 bytes

__device__ __forceinline__ v8f wmma_f16(v16h a, v16h b, v8f c) {
  return __builtin_amdgcn_wmma_f32_16x16x32_f16(false, a, false, b, (short)0, c,
                                                false, false);
}

union BB { v16h v; v8h p[2]; };

__global__ __launch_bounds__(NWAVES * 32, 1)
void moe_fused_kernel(const float* __restrict__ x,  const float* __restrict__ Wr,
                      const float* __restrict__ br, const float* __restrict__ W1,
                      const float* __restrict__ b1, const float* __restrict__ W2,
                      const float* __restrict__ b2, float* __restrict__ out) {
  extern __shared__ char smem[];
  _Float16* sA   = (_Float16*)smem;                 // [4][32 frag][32][16]
  _Float16* sW   = (_Float16*)(smem + SA_BYTES);    // [128 frag][32][16]
  _Float16* sAct = (_Float16*)(smem + SACT_OFF);    // [4][8 frag][32][16]
  float*    sP   = (float*)(smem + SP_OFF);         // [4][16][8]
  _Float16* sB2  = (_Float16*)(smem + SB2_OFF);     // [1024 col][8 k]
  _Float16* sWr  = (_Float16*)(smem + SWR_OFF);     // [64 row][8 n][16]
  _Float16* zbuf = (_Float16*)(smem + SZ_OFF);      // 16 halves of zero

  const int tid  = threadIdx.x;
  const int wave = tid >> 5;
  const int lane = tid & 31;
  const int n16  = lane & 15;   // A-frag row m / B,D-frag column n
  const int g    = lane >> 4;   // lane group

  const int tile = blockIdx.x * NWAVES + wave;
  const int tok0 = tile * TILE_M;

  _Float16* myA   = sA + wave * SA_HALVES;
  _Float16* myAct = sAct + wave * SACT_HALVES;
  float*    myP   = sP + wave * (TILE_M * Em);

  // ======= Phase 0: stage x (A-frag image), Wr strip, b2 strip, zeros =======
  if (tid < 8) ((unsigned int*)zbuf)[tid] = 0u;
  // Wr (1024x8 f32) -> strip: element Wr[k][n] at [((k>>4)*8+n)*16 + (k&15)]
  for (int it = 0; it < (Dm * Em) / (NWAVES * 32); ++it) {  // 64 per thread
    int lin = it * (NWAVES * 32) + tid;                     // coalesced read
    int k = lin >> 3, n = lin & 7;
    sWr[((k >> 4) * 8 + n) * 16 + (k & 15)] = (_Float16)Wr[lin];
  }
  // b2 (8x1024 f32) -> strip: element b2[k][col] at [col*8 + k]
  for (int it = 0; it < (Em * Dm) / (NWAVES * 32); ++it) {  // 64 per thread
    int lin = it * (NWAVES * 32) + tid;                     // coalesced read
    sB2[(lin & (Dm - 1)) * 8 + (lin >> 10)] = (_Float16)b2[lin];
  }
  // x tile -> A-fragment image.  Element (m,kk): lane gg*16+m,
  //   dword i0 = (kk&15)>>1, half kk&1, gg = (kk>>4)&1.
  for (int r = 0; r < TILE_M; ++r) {
    const float* src = x + (size_t)(tok0 + r) * Dm;
#pragma unroll
    for (int it = 0; it < Dm / (32 * 4); ++it) {
      int c  = it * 128 + lane * 4;              // 4 consecutive kk -> 2 dwords
      v4f f  = __builtin_nontemporal_load((const v4f*)(src + c));
      int kc = c >> 5;
      int kk = c & 31;
      int gg = (kk >> 4) & 1;
      int i0 = (kk & 15) >> 1;
      v4h h;
      h[0] = (_Float16)f[0]; h[1] = (_Float16)f[1];
      h[2] = (_Float16)f[2]; h[3] = (_Float16)f[3];
      *(v4h*)(myA + ((kc * 32 + gg * 16 + r) * 16 + i0 * 2)) = h;  // ds_store_b64
    }
  }
  __syncthreads();   // sWr/sB2 visible to everyone

  // ======= Phase 1: router logits via WMMA from LDS strips, then softmax ====
  {
    // pointer-select: dead columns (n16>=8) read the zero fragment, stride 0
    const _Float16* wrbase = (n16 < Em) ? (sWr + (g * 8 + n16) * 16) : zbuf;
    const int       wrstep = (n16 < Em) ? 256 : 0;   // halves per K-chunk
    v8f accR = {0.f, 0.f, 0.f, 0.f, 0.f, 0.f, 0.f, 0.f};
#pragma unroll 4
    for (int kc = 0; kc < KC; ++kc) {
      v16h a  = *(const v16h*)(myA + (kc * 32 + lane) * 16);   // 2x ds_load_b128
      v16h bf = *(const v16h*)(wrbase + kc * wrstep);          // 2x ds_load_b128
      accR = wmma_f16(a, bf, accR);
    }
    if (n16 < Em) {
      float brv = br[n16];
#pragma unroll
      for (int v = 0; v < 8; ++v)
        myP[(v + 8 * g) * Em + n16] = accR[v] + brv;   // logits -> LDS
    }
  }
  __syncthreads();
  if (lane < TILE_M) {  // softmax over E=8 experts, one token per lane
    float l[Em]; float mx = -3.0e38f;
#pragma unroll
    for (int e = 0; e < Em; ++e) { l[e] = myP[lane * Em + e]; mx = fmaxf(mx, l[e]); }
    float s = 0.f;
#pragma unroll
    for (int e = 0; e < Em; ++e) { l[e] = __expf(l[e] - mx); s += l[e]; }
    float inv = 1.f / s;
#pragma unroll
    for (int e = 0; e < Em; ++e) myP[lane * Em + e] = l[e] * inv;
  }

  // ======= Phase 2: per expert up-proj + SwiGLU, router prob folded in ======
  for (int e = 0; e < Em; ++e) {
    __syncthreads();  // previous expert's sW fully consumed; probs published
    // Stage W1[e] (1024x64 f32) into B-fragment image:
    // element (k,c): frag (k>>5, c>>4), lane (k>>4 &1)*16 + (c&15),
    //   dword (k&15)>>1, half k&1.
    const float* w1e = W1 + (size_t)e * (Dm * H2m);
    for (int it = 0; it < (Dm * H2m) / (NWAVES * 32 * 4); ++it) {  // 128 iters
      int idx = (it * (NWAVES * 32) + tid) * 4;
      int k = idx >> 6;
      int c = idx & 63;
      v4f f = __builtin_nontemporal_load((const v4f*)(w1e + idx));
      int kc = k >> 5, kk = k & 31;
      int base = (((kc * 4 + (c >> 4)) * 32) + (kk >> 4) * 16 + (c & 15)) * 16
               + ((kk & 15) >> 1) * 2 + (kk & 1);
      sW[base]      = (_Float16)f[0];
      sW[base + 16] = (_Float16)f[1];
      sW[base + 32] = (_Float16)f[2];
      sW[base + 48] = (_Float16)f[3];
    }
    __syncthreads();

    // Warm L2 with the next expert's W1 while we compute.
    if (e + 1 < Em) {
      const char* nxt = (const char*)(W1 + (size_t)(e + 1) * (Dm * H2m));
#pragma unroll
      for (int pf = 0; pf < 16; ++pf)
        __builtin_prefetch(nxt + (size_t)(pf * (NWAVES * 32) + tid) * 128, 0, 0);
    }

    // H1 = x(16x1024) @ W1[e](1024x64): 32 K-chunks x 4 N-tiles, 4 indep chains.
    v8f acc[4];
#pragma unroll
    for (int nt = 0; nt < 4; ++nt) acc[nt] = (v8f){0.f,0.f,0.f,0.f,0.f,0.f,0.f,0.f};
#pragma unroll 4
    for (int kc = 0; kc < KC; ++kc) {
      v16h a = *(const v16h*)(myA + (kc * 32 + lane) * 16);
#pragma unroll
      for (int nt = 0; nt < 4; ++nt) {
        v16h b = *(const v16h*)(sW + ((kc * 4 + nt) * 32 + lane) * 16);
        acc[nt] = wmma_f16(a, b, acc[nt]);
      }
    }

    // bias, SwiGLU, fold router prob, store act' tile in A-fragment image.
    float b1v[4];
#pragma unroll
    for (int nt = 0; nt < 4; ++nt) b1v[nt] = b1[e * H2m + nt * 16 + n16];
    float pm[8];
#pragma unroll
    for (int v = 0; v < 8; ++v) pm[v] = myP[(v + 8 * g) * Em + e];
    _Float16* actE = myAct + e * (32 * 16);
#pragma unroll
    for (int nt = 0; nt < 2; ++nt) {
      int c  = nt * 16 + n16;                 // H-dim position (0..31)
      int gA = (c >> 4) & 1;
      int iA = (c & 15) >> 1;
      int hA = c & 1;
#pragma unroll
      for (int v = 0; v < 8; ++v) {
        float xp = acc[nt][v] + b1v[nt];          // value half (cols 0..31)
        float gt = acc[nt + 2][v] + b1v[nt + 2];  // gate half  (cols 32..63)
        float sig = 1.f / (1.f + __expf(-gt));    // silu(gt) = gt*sig
        float av = xp * gt * sig * pm[v];         // router prob folded in
        int m = v + 8 * g;
        actE[(gA * 16 + m) * 16 + iA * 2 + hA] = (_Float16)av;
      }
    }
  }

  // ======= Phase 3: out = sum_e act'_e @ W2_e  (+ P @ b2 bias matmul) =======
  v16h af[Em];
#pragma unroll
  for (int e = 0; e < Em; ++e)
    af[e] = *(const v16h*)(myAct + e * (32 * 16) + lane * 16);  // 2x ds_load_b128

  v16h pfr;   // P (16x8 probs) as A-fragment, K padded to 32 with zeros
#pragma unroll
  for (int i = 0; i < 8; ++i) {
    if (i < 4) {
      float p0 = myP[n16 * Em + 2 * i];
      float p1 = myP[n16 * Em + 2 * i + 1];
      bool live = (g == 0);                  // only lanes 0..15, dwords 0..3 live
      pfr[2 * i]     = (_Float16)(live ? p0 : 0.f);
      pfr[2 * i + 1] = (_Float16)(live ? p1 : 0.f);
    } else {
      pfr[2 * i]     = (_Float16)0.f;
      pfr[2 * i + 1] = (_Float16)0.f;
    }
  }

  // bias-fragment strip pointer (pointer-select, no per-element masking)
  const _Float16* b2base = (g == 0) ? (sB2 + n16 * 8) : zbuf;
  const int       b2step = (g == 0) ? (16 * 8) : 0;   // halves per nt2
  v8h z8 = {(_Float16)0.f, (_Float16)0.f, (_Float16)0.f, (_Float16)0.f,
            (_Float16)0.f, (_Float16)0.f, (_Float16)0.f, (_Float16)0.f};

  _Float16* sW2 = (_Float16*)smem;   // repurpose sA+sW (256KB) as W2 frag cache

  for (int hf = 0; hf < 2; ++hf) {   // two 512-column halves of D
    __syncthreads();                 // x/W1 frags (or previous half) consumed
    const int colbase = hf * (Dm / 2);
    // Stage W2[:, :, colbase..colbase+511] into fragment image:
    // [e][nt2r][32 lanes][16 halves]; element (e,k,c512).
    for (int it = 0; it < (Em * Hm * (Dm / 2)) / (NWAVES * 32 * 4); ++it) { // 256
      int lin  = (it * (NWAVES * 32) + tid) * 4;
      int c512 = lin & 511;
      int k    = (lin >> 9) & 31;
      int e    = lin >> 14;
      v4f f = __builtin_nontemporal_load(
                (const v4f*)(W2 + (size_t)e * (Hm * Dm) + k * Dm + colbase + c512));
      int base = (((e * 32 + (c512 >> 4)) * 32) + (k >> 4) * 16 + (c512 & 15)) * 16
               + ((k & 15) >> 1) * 2 + (k & 1);
      sW2[base]      = (_Float16)f[0];
      sW2[base + 16] = (_Float16)f[1];
      sW2[base + 32] = (_Float16)f[2];
      sW2[base + 48] = (_Float16)f[3];
    }
    __syncthreads();

    for (int nt2r = 0; nt2r < 32; nt2r += 2) {   // 2-wide for WMMA ILP
      int nt2a = hf * 32 + nt2r;                 // absolute nt2 for bias strip
      int col0 = colbase + nt2r * 16 + n16;
      int col1 = col0 + 16;
      BB bb0, bb1;                               // P@b2 bias fragments
      bb0.p[0] = *(const v8h*)(b2base + nt2a * b2step);        // ds_load_b128
      bb1.p[0] = *(const v8h*)(b2base + (nt2a + 1) * b2step);
      bb0.p[1] = z8;
      bb1.p[1] = z8;
      v8f acc0 = {0.f,0.f,0.f,0.f,0.f,0.f,0.f,0.f};
      v8f acc1 = {0.f,0.f,0.f,0.f,0.f,0.f,0.f,0.f};
      acc0 = wmma_f16(pfr, bb0.v, acc0);
      acc1 = wmma_f16(pfr, bb1.v, acc1);
#pragma unroll
      for (int e = 0; e < Em; ++e) {
        v16h bw0 = *(const v16h*)(sW2 + ((e * 32 + nt2r) * 32 + lane) * 16);
        v16h bw1 = *(const v16h*)(sW2 + ((e * 32 + nt2r + 1) * 32 + lane) * 16);
        acc0 = wmma_f16(af[e], bw0, acc0);
        acc1 = wmma_f16(af[e], bw1, acc1);
      }
#pragma unroll
      for (int v = 0; v < 8; ++v) {
        int m = v + 8 * g;
        __builtin_nontemporal_store(acc0[v], &out[(size_t)(tok0 + m) * Dm + col0]);
        __builtin_nontemporal_store(acc1[v], &out[(size_t)(tok0 + m) * Dm + col1]);
      }
    }
  }
}

extern "C" void kernel_launch(void* const* d_in, const int* in_sizes, int n_in,
                              void* d_out, int out_size, void* d_ws, size_t ws_size,
                              hipStream_t stream) {
  (void)in_sizes; (void)n_in; (void)out_size; (void)d_ws; (void)ws_size;
  const float* x  = (const float*)d_in[0];
  const float* Wr = (const float*)d_in[1];
  const float* br = (const float*)d_in[2];
  const float* W1 = (const float*)d_in[3];
  const float* b1 = (const float*)d_in[4];
  const float* W2 = (const float*)d_in[5];
  const float* b2 = (const float*)d_in[6];
  float* out = (float*)d_out;

  dim3 grid(NTOK / (TILE_M * NWAVES));   // 512 workgroups
  dim3 block(NWAVES * 32);               // 4 wave32 waves
  moe_fused_kernel<<<grid, block, LDS_TOTAL, stream>>>(x, Wr, br, W1, b1, W2, b2, out);
}
